// Mamba_83854941487254
// MI455X (gfx1250) — matrix operand order repaired and checked by reading, not compiled, not run on hardware
//
#include <hip/hip_runtime.h>
#include <hip/hip_bf16.h>

// ---------------------------------------------------------------------------
// Mamba forward, 4 layers, B=2 L=1024 Dm=1024 Di=2048 rk=64 N=16 dc=4
// GEMMs in bf16 WMMA (v_wmma_f32_16x16x32_bf16, fp32 accum), scan in fp32.
// Wave tile 32x64 (8 WMMA / 12 b128 loads per K-step) for 2x arithmetic
// intensity vs. 32x32; block tile 64x256 with 8 waves (2x4).
// ---------------------------------------------------------------------------

typedef __bf16 bf16x8  __attribute__((ext_vector_type(8)));
typedef __bf16 bf16x16 __attribute__((ext_vector_type(16)));
typedef float  v8f     __attribute__((ext_vector_type(8)));

#define NL   4
#define BB   2
#define LL   1024
#define DM   1024
#define DI   2048
#define RK   64
#define NS   16
#define DC   4
#define MM   (BB*LL)          // 2048 token rows

// ------------------------------- f32 -> bf16 --------------------------------
__global__ __launch_bounds__(256) void k_cvt_bf16(const float* __restrict__ in,
                                                  __bf16* __restrict__ out, int n) {
    int t = blockIdx.x * 256 + threadIdx.x;
    if (t < n) out[t] = (__bf16)in[t];
}

// ------------------------------- RMSNorm ------------------------------------
// one block (256 thr) per token row; output bf16 activations for WMMA.
__global__ __launch_bounds__(256) void k_rmsnorm(const float* __restrict__ x,
                                                 const float* __restrict__ nw,
                                                 __bf16* __restrict__ h) {
    __shared__ float red[8];
    int row = blockIdx.x;
    int tid = threadIdx.x;
    int lane = tid & 31, wave = tid >> 5;
    float v[4]; float ss = 0.f;
#pragma unroll
    for (int j = 0; j < 4; ++j) {
        v[j] = x[(size_t)row * DM + tid + 256 * j];
        ss += v[j] * v[j];
    }
#pragma unroll
    for (int o = 16; o > 0; o >>= 1) ss += __shfl_xor(ss, o, 32);
    if (lane == 0) red[wave] = ss;
    __syncthreads();
    if (tid == 0) {
        float t = 0.f;
#pragma unroll
        for (int w = 0; w < 8; ++w) t += red[w];
        red[0] = t;
    }
    __syncthreads();
    float scale = rsqrtf(red[0] * (1.0f / DM) + 1e-5f);
#pragma unroll
    for (int j = 0; j < 4; ++j) {
        int c = tid + 256 * j;
        h[(size_t)row * DM + c] = (__bf16)(v[j] * scale * nw[c]);
    }
}

// --------------------------- WMMA NT GEMM -----------------------------------
// C[M,N](f32) = A[M,K](bf16, row-major, lda) * B[N,K](bf16, row-major, ldb)^T
// Block: 256 thr = 8 waves in 2x4 grid; block tile 64(M) x 256(N);
// wave tile 32x64 = 2x4 WMMA 16x16 tiles; K step 32.
__global__ __launch_bounds__(256) void k_gemm_nt(const __bf16* __restrict__ A,
                                                 const __bf16* __restrict__ B,
                                                 float* __restrict__ C,
                                                 int M, int N, int K,
                                                 int lda, int ldb, int ldc,
                                                 int accum) {
    int lane = threadIdx.x & 31;
    int wave = threadIdx.x >> 5;
    int wm = wave >> 2;              // 0..1
    int wn = wave & 3;               // 0..3
    int m0 = blockIdx.y * 64 + wm * 32;
    int n0 = blockIdx.x * 256 + wn * 64;
    int row  = lane & 15;
    int half = lane >> 4;            // lane group 0/1

    v8f acc[2][4];
#pragma unroll
    for (int i = 0; i < 2; ++i)
#pragma unroll
        for (int j = 0; j < 4; ++j) acc[i][j] = (v8f)0.f;

    // per-lane base pointers (K=0 slab)
    const __bf16* aBase[2];
    const __bf16* bBase[4];
#pragma unroll
    for (int i = 0; i < 2; ++i) {
        int r = m0 + 16 * i + row; if (r >= M) r = M - 1;
        aBase[i] = A + (size_t)r * lda;
    }
#pragma unroll
    for (int j = 0; j < 4; ++j) {
        int c = n0 + 16 * j + row; if (c >= N) c = N - 1;
        bBase[j] = B + (size_t)c * ldb;
    }

    for (int k0 = 0; k0 < K; k0 += 32) {
        bf16x16 af[2], bf[4];
#pragma unroll
        for (int i = 0; i < 2; ++i) {
            // 16-bit A layout: lane<16 holds K [0..7]+[16..23], lane>=16 holds
            // K [8..15]+[24..31] of the 32-wide K slab.
            const __bf16* ap = aBase[i] + k0;
            bf16x8 lo = *(const bf16x8*)(ap + half * 8);
            bf16x8 hi = *(const bf16x8*)(ap + 16 + half * 8);
            af[i] = __builtin_shufflevector(lo, hi, 0, 1, 2, 3, 4, 5, 6, 7,
                                            8, 9, 10, 11, 12, 13, 14, 15);
        }
#pragma unroll
        for (int j = 0; j < 4; ++j) {
            // B layout: lanes 0-15 hold K [0..7]+[16..23] (N=lane), lanes 16-31 K [8..15]+[24..31].
            const __bf16* bp = bBase[j] + k0;
            bf16x8 blo = *(const bf16x8*)(bp + half * 8);
            bf16x8 bhi = *(const bf16x8*)(bp + 16 + half * 8);
            bf[j] = __builtin_shufflevector(blo, bhi, 0, 1, 2, 3, 4, 5, 6, 7,
                                            8, 9, 10, 11, 12, 13, 14, 15);
        }
        if (k0 + 32 < K) {
            __builtin_prefetch(aBase[0] + k0 + 32, 0, 0);
            __builtin_prefetch(bBase[0] + k0 + 32, 0, 0);
            __builtin_prefetch(bBase[2] + k0 + 32, 0, 0);
        }
#pragma unroll
        for (int i = 0; i < 2; ++i)
#pragma unroll
            for (int j = 0; j < 4; ++j)
                acc[i][j] = __builtin_amdgcn_wmma_f32_16x16x32_bf16(
                    false, af[i], false, bf[j], (short)0, acc[i][j],
                    false, false);
    }

    // D layout: lane holds N = lane&15; VGPR v holds M = v + 8*(lane>>4)
#pragma unroll
    for (int i = 0; i < 2; ++i)
#pragma unroll
        for (int j = 0; j < 4; ++j) {
            int col = n0 + 16 * j + row;
            if (col >= N) continue;
#pragma unroll
            for (int v = 0; v < 8; ++v) {
                int rg = m0 + 16 * i + v + 8 * half;
                if (rg < M) {
                    float* p = C + (size_t)rg * ldc + col;
                    if (accum) *p += acc[i][j][v];
                    else       *p  = acc[i][j][v];
                }
            }
        }
}

// ---------------------- causal depthwise conv + SiLU ------------------------
__global__ __launch_bounds__(256) void k_conv_silu(const float* __restrict__ xz,
                                                   const float* __restrict__ cw,
                                                   const float* __restrict__ cb,
                                                   float* __restrict__ xc,
                                                   __bf16* __restrict__ xcb) {
    int t = blockIdx.x * 256 + threadIdx.x;       // t in [0, MM*DI)
    int d = t & (DI - 1);
    int m = t >> 11;
    int l = m & (LL - 1);
    float s = cb[d];
#pragma unroll
    for (int k = 0; k < DC; ++k) {
        int lk = l - (DC - 1) + k;
        if (lk >= 0) s += xz[(size_t)(m - (DC - 1) + k) * (2 * DI) + d] * cw[d * DC + k];
    }
    s = s / (1.f + __expf(-s));                   // SiLU
    xc[t]  = s;
    xcb[t] = (__bf16)s;
}

// --------------------------- SSM scan + gate --------------------------------
// one thread per (batch,channel); state[NS] in regs; B/C broadcast via LDS.
__global__ __launch_bounds__(256) void k_scan(const float* __restrict__ dlin,
                                              const float* __restrict__ bdt,
                                              const float* __restrict__ xc,
                                              const float* __restrict__ xz,
                                              const float* __restrict__ dbc,
                                              const float* __restrict__ alog,
                                              const float* __restrict__ Dp,
                                              __bf16* __restrict__ yzb) {
    __shared__ float sBC[2 * NS];
    int bb = blockIdx.x >> 3;                         // batch
    int d  = (blockIdx.x & 7) * 256 + threadIdx.x;    // channel

    float a2[NS], st[NS];
#pragma unroll
    for (int n = 0; n < NS; ++n) {
        a2[n] = -__expf(alog[(size_t)d * NS + n]) * 1.44269504f;  // A * log2(e)
        st[n] = 0.f;
    }
    float Dv = Dp[d];
    float bd = bdt[d];

    for (int l = 0; l < LL; ++l) {
        int m = bb * LL + l;
        if (threadIdx.x < 2 * NS)
            sBC[threadIdx.x] = dbc[(size_t)m * (RK + 2 * NS) + RK + threadIdx.x];
        __syncthreads();

        float dl = dlin[(size_t)m * DI + d] + bd;
        float delta = (dl > 20.f) ? dl : log1pf(__expf(dl));      // softplus
        float xv = xc[(size_t)m * DI + d];
        float dx = delta * xv;
        float y = 0.f;
#pragma unroll
        for (int n = 0; n < NS; ++n) {
            st[n] = exp2f(delta * a2[n]) * st[n] + dx * sBC[n];
            y += st[n] * sBC[NS + n];
        }
        y += Dv * xv;
        float z = xz[(size_t)m * (2 * DI) + DI + d];
        float g = z / (1.f + __expf(-z));                          // SiLU gate
        yzb[(size_t)m * DI + d] = (__bf16)(y * g);
        __syncthreads();
    }
}

// ---------------------------------------------------------------------------
extern "C" void kernel_launch(void* const* d_in, const int* in_sizes, int n_in,
                              void* d_out, int out_size, void* d_ws, size_t ws_size,
                              hipStream_t stream) {
    const float* x      = (const float*)d_in[0];
    const float* norm_w = (const float*)d_in[1];
    const float* W_in   = (const float*)d_in[2];
    const float* conv_w = (const float*)d_in[3];
    const float* conv_b = (const float*)d_in[4];
    const float* W_x    = (const float*)d_in[5];
    const float* W_dt   = (const float*)d_in[6];
    const float* b_dt   = (const float*)d_in[7];
    const float* A_log  = (const float*)d_in[8];
    const float* Dp     = (const float*)d_in[9];
    const float* W_out  = (const float*)d_in[10];
    float* out = (float*)d_out;

    // scratch carve-out
    char* w = (char*)d_ws;
    auto carve = [&](size_t bytes) {
        void* p = (void*)w;
        w += (bytes + 255) & ~(size_t)255;
        return p;
    };
    __bf16* h_bf   = (__bf16*)carve((size_t)MM * DM * 2);
    __bf16* win_bf = (__bf16*)carve((size_t)2 * DI * DM * 2);
    float*  xz     = (float*) carve((size_t)MM * 2 * DI * 4);
    float*  xc     = (float*) carve((size_t)MM * DI * 4);
    __bf16* xc_bf  = (__bf16*)carve((size_t)MM * DI * 2);
    __bf16* wx_bf  = (__bf16*)carve((size_t)(RK + 2 * NS) * DI * 2);
    float*  dbc    = (float*) carve((size_t)MM * (RK + 2 * NS) * 4);
    __bf16* dbc_bf = (__bf16*)carve((size_t)MM * (RK + 2 * NS) * 2);
    __bf16* wdt_bf = (__bf16*)carve((size_t)DI * RK * 2);
    float*  dlin   = (float*) carve((size_t)MM * DI * 4);
    __bf16* wout_bf= (__bf16*)carve((size_t)DM * DI * 2);
    __bf16* yz_bf  = (__bf16*)carve((size_t)MM * DI * 2);

    // residual accumulator = x
    hipMemcpyAsync(out, x, (size_t)MM * DM * 4, hipMemcpyDeviceToDevice, stream);

    for (int i = 0; i < NL; ++i) {
        const float* nw_i  = norm_w + (size_t)i * DM;
        const float* win_i = W_in   + (size_t)i * 2 * DI * DM;
        const float* cw_i  = conv_w + (size_t)i * DI * DC;
        const float* cb_i  = conv_b + (size_t)i * DI;
        const float* wx_i  = W_x    + (size_t)i * (RK + 2 * NS) * DI;
        const float* wdt_i = W_dt   + (size_t)i * DI * RK;
        const float* bdt_i = b_dt   + (size_t)i * DI;
        const float* al_i  = A_log  + (size_t)i * DI * NS;
        const float* D_i   = Dp     + (size_t)i * DI;
        const float* wo_i  = W_out  + (size_t)i * DM * DI;

        // 1. RMSNorm -> bf16
        k_rmsnorm<<<MM, 256, 0, stream>>>(out, nw_i, h_bf);

        // 2. xz = h @ W_in^T   [2048,4096] = [2048,1024] x [4096,1024]^T
        k_cvt_bf16<<<(2 * DI * DM) / 256, 256, 0, stream>>>(win_i, win_bf, 2 * DI * DM);
        k_gemm_nt<<<dim3(2 * DI / 256, MM / 64), 256, 0, stream>>>(
            h_bf, win_bf, xz, MM, 2 * DI, DM, DM, DM, 2 * DI, 0);

        // 3. causal conv + SiLU
        k_conv_silu<<<(MM * DI) / 256, 256, 0, stream>>>(xz, cw_i, cb_i, xc, xc_bf);

        // 4. dbc = xc @ W_x^T  [2048,96] = [2048,2048] x [96,2048]^T
        k_cvt_bf16<<<((RK + 2 * NS) * DI) / 256, 256, 0, stream>>>(
            wx_i, wx_bf, (RK + 2 * NS) * DI);
        k_gemm_nt<<<dim3(1, MM / 64), 256, 0, stream>>>(
            xc_bf, wx_bf, dbc, MM, RK + 2 * NS, DI, DI, DI, RK + 2 * NS, 0);

        // 5. dlin = dbc[:, :64] @ W_dt^T  [2048,2048] = [2048,64] x [2048,64]^T
        k_cvt_bf16<<<(MM * (RK + 2 * NS)) / 256, 256, 0, stream>>>(
            dbc, dbc_bf, MM * (RK + 2 * NS));
        k_cvt_bf16<<<(DI * RK) / 256, 256, 0, stream>>>(wdt_i, wdt_bf, DI * RK);
        k_gemm_nt<<<dim3(DI / 256, MM / 64), 256, 0, stream>>>(
            dbc_bf, wdt_bf, dlin, MM, DI, RK, RK + 2 * NS, RK, DI, 0);

        // 6. softplus + selective scan + D skip + SiLU(z) gate -> yz bf16
        k_scan<<<BB * (DI / 256), 256, 0, stream>>>(
            dlin, bdt_i, xc, xz, dbc, al_i, D_i, yz_bf);

        // 7. out += yz @ W_out^T  [2048,1024] += [2048,2048] x [1024,2048]^T
        k_cvt_bf16<<<(DM * DI) / 256, 256, 0, stream>>>(wo_i, wout_bf, DM * DI);
        k_gemm_nt<<<dim3(DM / 256, MM / 64), 256, 0, stream>>>(
            yz_bf, wout_bf, out, MM, DM, DI, DI, DI, DM, 1);
    }
}
